// SimpleRNN_n_16252156248546
// MI455X (gfx1250) — matrix operand order repaired and checked by reading, not compiled, not run on hardware
//
#include <hip/hip_runtime.h>

typedef __attribute__((ext_vector_type(16))) __bf16 v16bf;
typedef __attribute__((ext_vector_type(8)))  __bf16 v8bf;
typedef __attribute__((ext_vector_type(8)))  float  v8f;

#define T_SEQ   2048
#define BATCH   64
#define IN_F    512
#define HID_F   512
#define M_TOTAL (T_SEQ * BATCH)   // 131072 rows in the big GEMM

#define NTILES  32                // column tiles per weight matrix (512/16)
#define KBLKS   16                // k blocks per weight matrix (512/32)
#define FRAGS_PER_W (NTILES * KBLKS * 32)       // 16384 lane-fragments
#define W_SW_ELEMS  (FRAGS_PER_W * 16)          // 262144 bf16 per matrix

// ---- fragment helpers -------------------------------------------------------

static __device__ __forceinline__ void cvt4(v16bf& d, int base, float4 f) {
  d[base + 0] = (__bf16)f.x;
  d[base + 1] = (__bf16)f.y;
  d[base + 2] = (__bf16)f.z;
  d[base + 3] = (__bf16)f.w;
}

// B fragment (32x16, bf16): lane = column (lane&15); lane<16 holds K=kk..kk+15,
// lane>=16 holds K=kk+16..kk+31 -> 16 consecutive floats of one weight row.
static __device__ __forceinline__ v16bf b_frag_f32(const float* __restrict__ p) {
  const float4* q = (const float4*)p;
  float4 f0 = q[0], f1 = q[1], f2 = q[2], f3 = q[3];
  v16bf b;
  cvt4(b, 0, f0); cvt4(b, 4, f1); cvt4(b, 8, f2); cvt4(b, 12, f3);
  return b;
}

// A fragment (16x32, bf16) from fp32 global row: lane<16 holds
// K = 0..15, lane>=16 holds K = 16..31 (contiguous, matches B).
static __device__ __forceinline__ v16bf a_frag_global(const float* __restrict__ rowk,
                                                      int lane) {
  const int c0 = (lane & 16) ? 16 : 0;
  const float4* q0 = (const float4*)(rowk + c0);
  const float4* q1 = (const float4*)(rowk + 8 + c0);
  float4 f0 = q0[0], f1 = q0[1], f2 = q1[0], f3 = q1[1];
  v16bf a;
  cvt4(a, 0, f0); cvt4(a, 4, f1); cvt4(a, 8, f2); cvt4(a, 12, f3);
  return a;
}

// A fragment (16x32, bf16) from LDS-resident bf16 hidden state (two 16B runs).
static __device__ __forceinline__ v16bf a_frag_lds(const __bf16* rowk, int lane) {
  const int c0 = (lane & 16) ? 16 : 0;
  v8bf r0 = *(const v8bf*)(rowk + c0);
  v8bf r1 = *(const v8bf*)(rowk + 8 + c0);
  return __builtin_shufflevector(r0, r1, 0, 1, 2, 3, 4, 5, 6, 7,
                                 8, 9, 10, 11, 12, 13, 14, 15);
}

// ---- Phase 0: pre-swizzle both weight matrices into bf16 B-fragment order ---
// ws layout: frag index g = which*16384 + (nt*16 + kb)*32 + lane, 16 bf16 each.

__global__ __launch_bounds__(256) void swizzle_w_kernel(
    const float* __restrict__ wih, const float* __restrict__ whh,
    __bf16* __restrict__ wsw) {
  const int gid   = blockIdx.x * 256 + threadIdx.x; // 0..32767
  const int which = gid >> 14;                      // 0: W_ih, 1: W_hh
  const int rem   = gid & 16383;
  const int lane  = rem & 31;
  const int tk    = rem >> 5;                       // nt*16 + kb
  const int nt    = tk >> 4;
  const int kb    = tk & 15;
  const float* w  = which ? whh : wih;
  const int col   = nt * 16 + (lane & 15);
  const int kr    = kb * 32 + ((lane & 16) ? 16 : 0);
  v16bf b = b_frag_f32(w + (size_t)col * 512 + kr);
  *(v16bf*)(wsw + (size_t)gid * 16) = b;
}

// ---- Phase 1: xi[t,b,h] = x @ W_ih^T + (b_ih + b_hh) ------------------------
// Grid: (M_TOTAL/64, HID/256), block 256 (8 waves). Wave tile = 32(M) x 64(N).

template <bool SW>
__global__ __launch_bounds__(256) void xi_gemm_kernel(
    const float* __restrict__ x, const float* __restrict__ wih,
    const __bf16* __restrict__ wsw,
    const float* __restrict__ bih, const float* __restrict__ bhh,
    float* __restrict__ out) {
  const int lane = threadIdx.x & 31;
  const int wave = threadIdx.x >> 5;
  const int wm = wave >> 2;                       // 0..1
  const int wn = wave & 3;                        // 0..3
  const int rowBase = blockIdx.x * 64 + wm * 32;  // M
  const int colBase = blockIdx.y * 256 + wn * 64; // N (hidden dim)
  const int ntBase  = colBase >> 4;               // first of 4 column tiles
  const int ln15 = lane & 15;
  const int mrow = (lane & 16) ? 8 : 0;

  v8f acc[2][4];
#pragma unroll
  for (int i = 0; i < 2; ++i)
#pragma unroll
    for (int j = 0; j < 4; ++j)
      acc[i][j] = (v8f){0.f, 0.f, 0.f, 0.f, 0.f, 0.f, 0.f, 0.f};

  for (int kk = 0; kk < IN_F; kk += 32) {
    const int kb = kk >> 5;
    v16bf a[2];
#pragma unroll
    for (int mi = 0; mi < 2; ++mi) {
      const int row = rowBase + mi * 16 + ln15;
      a[mi] = a_frag_global(x + (size_t)row * IN_F + kk, lane);
    }
    const int kr = kk + ((lane & 16) ? 16 : 0);
#pragma unroll
    for (int ni = 0; ni < 4; ++ni) {
      v16bf b;
      if constexpr (SW) {
        b = *(const v16bf*)(wsw +
            (((size_t)(ntBase + ni) * 16 + kb) * 32 + lane) * 16);
      } else {
        const int col = colBase + ni * 16 + ln15;
        b = b_frag_f32(wih + (size_t)col * IN_F + kr);
      }
#pragma unroll
      for (int mi = 0; mi < 2; ++mi)
        acc[mi][ni] = __builtin_amdgcn_wmma_f32_16x16x32_bf16(
            false, a[mi], false, b, (short)0, acc[mi][ni], false, false);
    }
  }

#pragma unroll
  for (int ni = 0; ni < 4; ++ni) {
    const int col = colBase + ni * 16 + ln15;
    const float bias = bih[col] + bhh[col];
#pragma unroll
    for (int mi = 0; mi < 2; ++mi) {
#pragma unroll
      for (int r = 0; r < 8; ++r) {
        const int row = rowBase + mi * 16 + mrow + r;
        out[(size_t)row * HID_F + col] = acc[mi][ni][r] + bias;
      }
    }
  }
}

// ---- Phase 2: sequential scan, single persistent workgroup ------------------
// h_t = relu(xi_t + h_{t-1} @ W_hh^T). Ping-pong hidden state in LDS (2x64KB
// bf16 of the 320KB WGP LDS) -> single s_barrier per step. 32 waves, wave w
// owns output columns 16w..16w+15; its pre-swizzled W_hh slice (16KB bf16)
// streams from L2 with zero conversion VALU. Next step's xi rows are pulled
// ahead with global_prefetch (no LOADcnt) so the epilogue loads hit cache.

template <bool SW>
__global__ __launch_bounds__(1024) void rnn_scan_kernel(
    const float* __restrict__ whh, const __bf16* __restrict__ wsw,
    float* __restrict__ out) {
  __shared__ __align__(16) __bf16 hbuf[2][BATCH * HID_F]; // 2 x 64KB

  const int tid = threadIdx.x;
  const int lane = tid & 31;
  const int wave = tid >> 5;       // 0..31 -> column tile
  const int ln15 = lane & 15;
  const int col = wave * 16 + ln15;
  const int mrow = (lane & 16) ? 8 : 0;
  const float* wrow = whh + (size_t)col * HID_F;          // fallback path
  const __bf16* wfrag0 = SW ? (wsw + (((size_t)wave * 16) * 32 + lane) * 16)
                            : nullptr;                    // frag for kb=0

  // h0 = 0 (buffer 0 is read at t=0)
  for (int i = tid; i < BATCH * HID_F; i += 1024) hbuf[0][i] = (__bf16)0.f;
  __syncthreads();

  for (int t = 0; t < T_SEQ; ++t) {
    const __bf16* hrd = hbuf[t & 1];
    __bf16* hwr = hbuf[(t & 1) ^ 1];
    float* xt = out + (size_t)t * (BATCH * HID_F);

    // Warm next step's xi slice: 2 rows per lane x 32 lanes = 64 rows of this
    // wave's 16-column segment. Prefetch has no counter -> doesn't perturb the
    // k-loop's load pipelining.
    if (t + 1 < T_SEQ) {
      const float* xn = xt + BATCH * HID_F;
      __builtin_prefetch(xn + (size_t)(lane * 2 + 0) * HID_F + wave * 16, 0, 1);
      __builtin_prefetch(xn + (size_t)(lane * 2 + 1) * HID_F + wave * 16, 0, 1);
    }

    v8f acc[4];
#pragma unroll
    for (int m0 = 0; m0 < 4; ++m0)
      acc[m0] = (v8f){0.f, 0.f, 0.f, 0.f, 0.f, 0.f, 0.f, 0.f};

    for (int kk = 0; kk < HID_F; kk += 32) {
      const int kb = kk >> 5;
      v16bf b;
      if constexpr (SW) {
        b = *(const v16bf*)(wfrag0 + (size_t)kb * (32 * 16));
      } else {
        const int kr = kk + ((lane & 16) ? 16 : 0);
        b = b_frag_f32(wrow + kr);
      }
#pragma unroll
      for (int m0 = 0; m0 < 4; ++m0) {
        v16bf a = a_frag_lds(&hrd[(m0 * 16 + ln15) * HID_F + kk], lane);
        acc[m0] = __builtin_amdgcn_wmma_f32_16x16x32_bf16(
            false, a, false, b, (short)0, acc[m0], false, false);
      }
    }

    // No barrier needed before the update: h_t goes to the *other* buffer.
#pragma unroll
    for (int m0 = 0; m0 < 4; ++m0) {
#pragma unroll
      for (int r = 0; r < 8; ++r) {
        const int row = m0 * 16 + mrow + r;           // batch index
        const size_t idx = (size_t)row * HID_F + col; // [b, h]
        float v = acc[m0][r] + xt[idx];
        v = v > 0.f ? v : 0.f;
        xt[idx] = v;                                  // output[t]
        hwr[idx] = (__bf16)v;                         // state for t+1
        if (t == T_SEQ - 1)
          out[(size_t)T_SEQ * BATCH * HID_F + idx] = v; // hx tail
      }
    }
    __syncthreads(); // swap buffers: h_t visible before step t+1 reads it
  }
}

// ---- launcher ---------------------------------------------------------------

extern "C" void kernel_launch(void* const* d_in, const int* in_sizes, int n_in,
                              void* d_out, int out_size, void* d_ws, size_t ws_size,
                              hipStream_t stream) {
  (void)in_sizes; (void)n_in; (void)out_size;
  const float* x   = (const float*)d_in[0];
  const float* wih = (const float*)d_in[1];
  const float* whh = (const float*)d_in[2];
  const float* bih = (const float*)d_in[3];
  const float* bhh = (const float*)d_in[4];
  float* out = (float*)d_out;

  const size_t need = (size_t)2 * W_SW_ELEMS * sizeof(__bf16); // 1 MB
  dim3 g1(M_TOTAL / 64, HID_F / 256);

  if (d_ws != nullptr && ws_size >= need) {
    __bf16* wsw = (__bf16*)d_ws;
    swizzle_w_kernel<<<(2 * FRAGS_PER_W) / 256, 256, 0, stream>>>(wih, whh, wsw);
    xi_gemm_kernel<true><<<g1, 256, 0, stream>>>(x, wih, wsw, bih, bhh, out);
    rnn_scan_kernel<true><<<1, 1024, 0, stream>>>(whh, wsw + W_SW_ELEMS, out);
  } else {
    xi_gemm_kernel<false><<<g1, 256, 0, stream>>>(x, wih, nullptr, bih, bhh, out);
    rnn_scan_kernel<false><<<1, 1024, 0, stream>>>(whh, nullptr, out);
  }
}
